// DecoderBlock_33672543601183
// MI455X (gfx1250) — matrix operand order repaired and checked by reading, not compiled, not run on hardware
//
#include <hip/hip_runtime.h>
#include <hip/hip_bf16.h>

// ---------------------------------------------------------------------------
// Transformer decoder block (pre-norm attn + FFN) for gfx1250 (MI455X).
// fp32 -> bf16 once, all matmuls via v_wmma_f32_16x16x32_bf16, f32 accumulate.
// GEMM: 32x64 tile per wave, software-pipelined (double-buffered fragments).
// Attention: flash-style online softmax, LDS re-fragmentation for P*V.
// ---------------------------------------------------------------------------

typedef __bf16          v16bf   __attribute__((ext_vector_type(16)));
typedef float           v8f     __attribute__((ext_vector_type(8)));
typedef unsigned short  ushort8v __attribute__((ext_vector_type(8)));

#define B_   2
#define S_   2048
#define D_   1024
#define H_   16
#define DH_  64
#define DFF_ 4096
#define M_   (B_ * S_)   // 4096 rows for all token-major GEMMs

__device__ __forceinline__ unsigned short f2bf(float f) {
    unsigned int u = __float_as_uint(f);
    u += 0x7FFFu + ((u >> 16) & 1u);     // round-to-nearest-even
    return (unsigned short)(u >> 16);
}

union Frag16 {                            // one wave-lane's 16 bf16 operand values
    ushort8v h[2];
    v16bf    v;
};

__device__ __forceinline__ v8f wmma_bf16(v16bf a, v16bf b, v8f c) {
    // D = A(16x32 bf16) * B(32x16 bf16) + C(16x16 f32)
    return __builtin_amdgcn_wmma_f32_16x16x32_bf16(
        /*neg_a=*/false, a, /*neg_b=*/false, b,
        /*c_mod=*/(short)0, c, /*reuse_a=*/false, /*reuse_b=*/false);
}

// ---------------------------------------------------------------------------
// fp32 W[K][N]  ->  bf16 Wt[N][K]  (so WMMA B-fragments are contiguous loads)
// ---------------------------------------------------------------------------
__global__ void wt_bf16_kernel(const float* __restrict__ W,
                               unsigned short* __restrict__ Wt,
                               int K, int N) {
    long long t = (long long)blockIdx.x * blockDim.x + threadIdx.x;
    long long total = (long long)K * N;
    if (t >= total) return;
    int k = (int)(t / N);
    int n = (int)(t % N);
    Wt[(long long)n * K + k] = f2bf(W[t]);
}

// ---------------------------------------------------------------------------
// LayerNorm: fp32 X[row][D_] -> bf16 out[row][D_], one 256-thread block / row
// ---------------------------------------------------------------------------
__global__ __launch_bounds__(256) void ln_bf16_kernel(
        const float* __restrict__ X, const float* __restrict__ g,
        const float* __restrict__ be, unsigned short* __restrict__ out) {
    __shared__ float red[256];
    __shared__ float mu_s, rs_s;
    const int tid = threadIdx.x;
    const float* x = X + (long long)blockIdx.x * D_;

    float acc = 0.f;
#pragma unroll
    for (int i = 0; i < D_ / 256; ++i) acc += x[tid + 256 * i];
    red[tid] = acc; __syncthreads();
    for (int off = 128; off > 0; off >>= 1) {
        if (tid < off) red[tid] += red[tid + off];
        __syncthreads();
    }
    if (tid == 0) mu_s = red[0] * (1.f / D_);
    __syncthreads();
    const float mu = mu_s;

    float vacc = 0.f;
#pragma unroll
    for (int i = 0; i < D_ / 256; ++i) {
        float d = x[tid + 256 * i] - mu;
        vacc += d * d;
    }
    __syncthreads();
    red[tid] = vacc; __syncthreads();
    for (int off = 128; off > 0; off >>= 1) {
        if (tid < off) red[tid] += red[tid + off];
        __syncthreads();
    }
    if (tid == 0) rs_s = rsqrtf(red[0] * (1.f / D_) + 1e-5f);
    __syncthreads();
    const float rs = rs_s;

    unsigned short* o = out + (long long)blockIdx.x * D_;
#pragma unroll
    for (int i = 0; i < D_ / 256; ++i) {
        int c = tid + 256 * i;
        o[c] = f2bf((x[c] - mu) * rs * g[c] + be[c]);
    }
}

// ---------------------------------------------------------------------------
// V[b*S+s][h*DH+d] -> Vt[((b*H+h)*DH+d)][s]   (bf16, for P*V B-fragments)
// ---------------------------------------------------------------------------
__global__ void vtrans_kernel(const unsigned short* __restrict__ V,
                              unsigned short* __restrict__ Vt) {
    long long t = (long long)blockIdx.x * blockDim.x + threadIdx.x; // over M_*D_
    int row = (int)(t / D_);
    int col = (int)(t % D_);
    int b = row / S_, s = row % S_;
    int h = col / DH_, d = col % DH_;
    Vt[(((long long)(b * H_ + h) * DH_ + d) * S_) + s] = V[t];
}

// ---------------------------------------------------------------------------
// WMMA GEMM: C[M][N] = A[M][K](bf16) * Bt[N][K](bf16)^T (+bias)(+relu)(+resid)
// 4 waves / block; wave tile 32x64 (2 A-frags x 4 B-frags = 8 WMMA / K-step).
// Software-pipelined: next K-step fragments load while current WMMAs issue.
// ---------------------------------------------------------------------------
__global__ __launch_bounds__(128) void gemm_wmma_kernel(
        const unsigned short* __restrict__ A,
        const unsigned short* __restrict__ Bt,
        const float* __restrict__ bias, const float* __restrict__ resid,
        float* __restrict__ outF, unsigned short* __restrict__ outH,
        int M, int N, int K, int relu) {
    const int wave = threadIdx.x >> 5;
    const int lane = threadIdx.x & 31;
    const int lo = lane & 15;         // column / row-in-fragment lane id
    const int hi = lane >> 4;         // K-chunk select
    const int mbase = (blockIdx.y * 4 + wave) * 32;
    const int nbase = blockIdx.x * 64;

    // A fragments: lane holds row (mbase[+16]+lo), K pieces [hi*8..+7], +16
    const unsigned short* arow0 = A + (long long)(mbase + lo) * K + hi * 8;
    const unsigned short* arow1 = arow0 + (long long)16 * K;
    // B fragments: lane holds col (nbase+t*16+lo), K pieces [hi*8..+7], +16
    const unsigned short* bb = Bt + (long long)(nbase + lo) * K + hi * 8;
    const long long bstep = (long long)16 * K;   // between n sub-tiles

    v8f acc[2][4] = {};
    Frag16 a0, a1, b[4];

    // prologue: fragments for k = 0
    a0.h[0] = *(const ushort8v*)(arow0);
    a0.h[1] = *(const ushort8v*)(arow0 + 16);
    a1.h[0] = *(const ushort8v*)(arow1);
    a1.h[1] = *(const ushort8v*)(arow1 + 16);
#pragma unroll
    for (int t = 0; t < 4; ++t) {
        b[t].h[0] = *(const ushort8v*)(bb + t * bstep);
        b[t].h[1] = *(const ushort8v*)(bb + t * bstep + 16);
    }

    for (int k = 32; k < K; k += 32) {
        // issue next-step loads first (consumed only after this step's WMMAs)
        Frag16 na0, na1, nb[4];
        na0.h[0] = *(const ushort8v*)(arow0 + k);
        na0.h[1] = *(const ushort8v*)(arow0 + k + 16);
        na1.h[0] = *(const ushort8v*)(arow1 + k);
        na1.h[1] = *(const ushort8v*)(arow1 + k + 16);
#pragma unroll
        for (int t = 0; t < 4; ++t) {
            nb[t].h[0] = *(const ushort8v*)(bb + t * bstep + k);
            nb[t].h[1] = *(const ushort8v*)(bb + t * bstep + k + 16);
        }
        if (k + 96 < K) __builtin_prefetch(arow0 + k + 96, 0, 3);

        // 8 WMMAs on the current (already-resident) fragments
#pragma unroll
        for (int t = 0; t < 4; ++t) acc[0][t] = wmma_bf16(a0.v, b[t].v, acc[0][t]);
#pragma unroll
        for (int t = 0; t < 4; ++t) acc[1][t] = wmma_bf16(a1.v, b[t].v, acc[1][t]);

        a0 = na0; a1 = na1;
#pragma unroll
        for (int t = 0; t < 4; ++t) b[t] = nb[t];
    }
    // epilogue K-step
#pragma unroll
    for (int t = 0; t < 4; ++t) acc[0][t] = wmma_bf16(a0.v, b[t].v, acc[0][t]);
#pragma unroll
    for (int t = 0; t < 4; ++t) acc[1][t] = wmma_bf16(a1.v, b[t].v, acc[1][t]);

#pragma unroll
    for (int mf = 0; mf < 2; ++mf) {
#pragma unroll
        for (int r = 0; r < 8; ++r) {
            const int m = mbase + mf * 16 + hi * 8 + r;  // C: m = 8*(lane>>4)+vgpr
#pragma unroll
            for (int t = 0; t < 4; ++t) {
                const int n = nbase + t * 16 + lo;
                float v = acc[mf][t][r];
                if (bias)  v += bias[n];
                if (relu)  v = fmaxf(v, 0.f);
                if (resid) v += resid[(long long)m * N + n];
                if (outF)  outF[(long long)m * N + n] = v;
                if (outH)  outH[(long long)m * N + n] = f2bf(v);
            }
        }
    }
}

// ---------------------------------------------------------------------------
// Flash attention: one wave per (b, h, 16-query tile). Causal, online softmax.
// scores = Q*K^T / 8 ; P staged bf16 through LDS to re-fragment; O += P*V.
// ---------------------------------------------------------------------------
__global__ __launch_bounds__(128) void attn_kernel(
        const unsigned short* __restrict__ Q,
        const unsigned short* __restrict__ Kb,
        const unsigned short* __restrict__ Vt,
        unsigned short* __restrict__ Ctx) {
    __shared__ __align__(16) unsigned short plds[4][16 * 32];
    const int wave = threadIdx.x >> 5;
    const int lane = threadIdx.x & 31;
    const int lo = lane & 15;
    const int hi = lane >> 4;

    const int wid = blockIdx.x * 4 + wave;       // 0 .. B_*H_*(S_/16)-1
    const int b  = wid >> 11;                    // H_*(S_/16) = 2048
    const int h  = (wid >> 7) & (H_ - 1);
    const int qt = wid & 127;
    unsigned short* myl = plds[wave];

    const unsigned short* qrow =
        Q + (long long)(b * S_ + qt * 16 + lo) * D_ + h * DH_ + hi * 8;

    // Q fragments are loop-invariant: load once
    Frag16 qf[2];
#pragma unroll
    for (int kk = 0; kk < 2; ++kk) {
        qf[kk].h[0] = *(const ushort8v*)(qrow + kk * 32);
        qf[kk].h[1] = *(const ushort8v*)(qrow + kk * 32 + 16);
    }

    v8f o[4] = {};
    float m_r[8], l_r[8];
#pragma unroll
    for (int r = 0; r < 8; ++r) { m_r[r] = -1e30f; l_r[r] = 0.f; }

    const int nkt = (qt * 16 + 15) / 32 + 1;     // causal: key tiles of 32
    for (int kt = 0; kt < nkt; ++kt) {
        const int kbase = kt * 32;

        // ---- scores: 16 queries x 32 keys, two 16x16 WMMA column halves
        v8f s[2] = {};
#pragma unroll
        for (int half = 0; half < 2; ++half) {
            const unsigned short* kp =
                Kb + (long long)(b * S_ + kbase + half * 16 + lo) * D_ +
                h * DH_ + hi * 8;
#pragma unroll
            for (int kk = 0; kk < 2; ++kk) {
                Frag16 bf;
                bf.h[0] = *(const ushort8v*)(kp + kk * 32);
                bf.h[1] = *(const ushort8v*)(kp + kk * 32 + 16);
                s[half] = wmma_bf16(qf[kk].v, bf.v, s[half]);
            }
        }

        // ---- online softmax update (rows live in 16-lane groups)
#pragma unroll
        for (int r = 0; r < 8; ++r) {
            const int qi = qt * 16 + hi * 8 + r;
            float s0 = s[0][r] * 0.125f;                  // 1/sqrt(64)
            if (kbase + lo > qi)      s0 = -1e30f;        // causal mask
            float s1 = s[1][r] * 0.125f;
            if (kbase + 16 + lo > qi) s1 = -1e30f;

            float tm = fmaxf(s0, s1);
#pragma unroll
            for (int off = 1; off < 16; off <<= 1)
                tm = fmaxf(tm, __shfl_xor(tm, off, 32));
            const float newm  = fmaxf(m_r[r], tm);
            const float alpha = __expf(m_r[r] - newm);
            m_r[r] = newm;
            const float p0 = __expf(s0 - newm);
            const float p1 = __expf(s1 - newm);
            float rs = p0 + p1;
#pragma unroll
            for (int off = 1; off < 16; off <<= 1)
                rs += __shfl_xor(rs, off, 32);
            l_r[r] = l_r[r] * alpha + rs;
#pragma unroll
            for (int t = 0; t < 4; ++t) o[t][r] *= alpha;

            const int mrow = hi * 8 + r;                  // C-layout row
            myl[mrow * 32 + lo]      = f2bf(p0);
            myl[mrow * 32 + 16 + lo] = f2bf(p1);
        }

        // re-fragment P from C layout to A layout via LDS (same-wave DS order)
        asm volatile("s_wait_dscnt 0" ::: "memory");
        Frag16 p;
        p.h[0] = *(const ushort8v*)(myl + lo * 32 + hi * 8);
        p.h[1] = *(const ushort8v*)(myl + lo * 32 + hi * 8 + 16);

        // ---- O += P(16x32) * V(32x16) per 16-wide d sub-tile
#pragma unroll
        for (int t = 0; t < 4; ++t) {
            const unsigned short* vp =
                Vt + ((long long)((b * H_ + h) * DH_ + t * 16 + lo)) * S_ +
                kbase + hi * 8;
            Frag16 vf;
            vf.h[0] = *(const ushort8v*)(vp);
            vf.h[1] = *(const ushort8v*)(vp + 16);
            o[t] = wmma_bf16(p.v, vf.v, o[t]);
        }
        asm volatile("" ::: "memory");
    }

#pragma unroll
    for (int r = 0; r < 8; ++r) {
        const float inv = 1.f / l_r[r];
        const long long row = (long long)(b * S_ + qt * 16 + hi * 8 + r);
#pragma unroll
        for (int t = 0; t < 4; ++t)
            Ctx[row * D_ + h * DH_ + t * 16 + lo] = f2bf(o[t][r] * inv);
    }
}

// ---------------------------------------------------------------------------
// Orchestration
// ---------------------------------------------------------------------------
extern "C" void kernel_launch(void* const* d_in, const int* in_sizes, int n_in,
                              void* d_out, int out_size, void* d_ws, size_t ws_size,
                              hipStream_t stream) {
    const float* x   = (const float*)d_in[0];
    // d_in[1] = causal mask (bool) — structure is known, not needed
    const float* Wq  = (const float*)d_in[2];
    const float* Wk  = (const float*)d_in[3];
    const float* Wv  = (const float*)d_in[4];
    const float* Wo  = (const float*)d_in[5];
    const float* w1  = (const float*)d_in[6];
    const float* b1  = (const float*)d_in[7];
    const float* w2  = (const float*)d_in[8];
    const float* b2  = (const float*)d_in[9];
    const float* g1  = (const float*)d_in[10];
    const float* be1 = (const float*)d_in[11];
    const float* g2  = (const float*)d_in[12];
    const float* be2 = (const float*)d_in[13];

    char* p = (char*)d_ws;
    auto carve = [&](size_t bytes) -> void* {
        void* r = (void*)p;
        p += (bytes + 255) & ~(size_t)255;
        return r;
    };
    unsigned short* wqT = (unsigned short*)carve((size_t)D_ * D_ * 2);
    unsigned short* wkT = (unsigned short*)carve((size_t)D_ * D_ * 2);
    unsigned short* wvT = (unsigned short*)carve((size_t)D_ * D_ * 2);
    unsigned short* woT = (unsigned short*)carve((size_t)D_ * D_ * 2);
    unsigned short* w1T = (unsigned short*)carve((size_t)D_ * DFF_ * 2);
    unsigned short* w2T = (unsigned short*)carve((size_t)DFF_ * D_ * 2);
    unsigned short* xn1 = (unsigned short*)carve((size_t)M_ * D_ * 2);
    unsigned short* qb  = (unsigned short*)carve((size_t)M_ * D_ * 2);
    unsigned short* kb  = (unsigned short*)carve((size_t)M_ * D_ * 2);
    unsigned short* vb  = (unsigned short*)carve((size_t)M_ * D_ * 2);
    unsigned short* vt  = (unsigned short*)carve((size_t)M_ * D_ * 2);
    unsigned short* ctx = (unsigned short*)carve((size_t)M_ * D_ * 2);
    float*          hbf = (float*)carve((size_t)M_ * D_ * 4);
    unsigned short* hn  = (unsigned short*)carve((size_t)M_ * D_ * 2);
    unsigned short* a1  = (unsigned short*)carve((size_t)M_ * DFF_ * 2);

    const dim3 blk256(256);
    const dim3 blk128(128);

    // weights -> bf16 transposed
    wt_bf16_kernel<<<dim3((D_ * D_ + 255) / 256), blk256, 0, stream>>>(Wq, wqT, D_, D_);
    wt_bf16_kernel<<<dim3((D_ * D_ + 255) / 256), blk256, 0, stream>>>(Wk, wkT, D_, D_);
    wt_bf16_kernel<<<dim3((D_ * D_ + 255) / 256), blk256, 0, stream>>>(Wv, wvT, D_, D_);
    wt_bf16_kernel<<<dim3((D_ * D_ + 255) / 256), blk256, 0, stream>>>(Wo, woT, D_, D_);
    wt_bf16_kernel<<<dim3((D_ * DFF_ + 255) / 256), blk256, 0, stream>>>(w1, w1T, D_, DFF_);
    wt_bf16_kernel<<<dim3((DFF_ * D_ + 255) / 256), blk256, 0, stream>>>(w2, w2T, DFF_, D_);

    // ln1
    ln_bf16_kernel<<<dim3(M_), blk256, 0, stream>>>(x, g1, be1, xn1);

    // Q, K, V projections  (grid: N/64 x M/128)
    gemm_wmma_kernel<<<dim3(D_ / 64, M_ / 128), blk128, 0, stream>>>(
        xn1, wqT, nullptr, nullptr, nullptr, qb, M_, D_, D_, 0);
    gemm_wmma_kernel<<<dim3(D_ / 64, M_ / 128), blk128, 0, stream>>>(
        xn1, wkT, nullptr, nullptr, nullptr, kb, M_, D_, D_, 0);
    gemm_wmma_kernel<<<dim3(D_ / 64, M_ / 128), blk128, 0, stream>>>(
        xn1, wvT, nullptr, nullptr, nullptr, vb, M_, D_, D_, 0);

    // V -> per-head transposed
    vtrans_kernel<<<dim3((M_ * D_) / 256), blk256, 0, stream>>>(vb, vt);

    // flash attention
    attn_kernel<<<dim3((B_ * H_ * (S_ / 16)) / 4), blk128, 0, stream>>>(qb, kb, vt, ctx);

    // O projection + residual 1 -> h (fp32)
    gemm_wmma_kernel<<<dim3(D_ / 64, M_ / 128), blk128, 0, stream>>>(
        ctx, woT, nullptr, x, hbf, nullptr, M_, D_, D_, 0);

    // ln2
    ln_bf16_kernel<<<dim3(M_), blk256, 0, stream>>>(hbf, g2, be2, hn);

    // FFN: relu(hn @ w1 + b1) -> a1 (bf16)
    gemm_wmma_kernel<<<dim3(DFF_ / 64, M_ / 128), blk128, 0, stream>>>(
        hn, w1T, b1, nullptr, nullptr, a1, M_, DFF_, D_, 1);

    // out = h + a1 @ w2 + b2 (fp32)
    gemm_wmma_kernel<<<dim3(D_ / 64, M_ / 128), blk128, 0, stream>>>(
        a1, w2T, b2, hbf, (float*)d_out, nullptr, M_, D_, DFF_, 0);
}
